// LaMDA_5806795784201
// MI455X (gfx1250) — matrix operand order repaired and checked by reading, not compiled, not run on hardware
//
#include <hip/hip_runtime.h>

// ---------------------------------------------------------------------------
// MI455X (gfx1250) POI-recommendation forward pass.
// Dense GEMMs: WMMA f16 (fp32 accum), LDS tiles staged by the Tensor Data
// Mover (tensor_load_to_lds, TENSORcnt) with double buffering so DMA overlaps
// the WMMA pipe. Softmax/LN/pool in fp32 VALU; GCN scatter via f32 atomics.
// ---------------------------------------------------------------------------

typedef __attribute__((ext_vector_type(16))) _Float16 v16h;
typedef __attribute__((ext_vector_type(8)))  float    v8f;
typedef unsigned int u32;
typedef u32 v4u __attribute__((ext_vector_type(4)));
typedef u32 v8u __attribute__((ext_vector_type(8)));
typedef unsigned long long u64;

#define NPOI 100000
#define DD   128
#define EE   900000
#define TT   32768
#define BB   512
#define LL   128

// ---------------- Tensor Data Mover: 2-D tile load (rows x 32 halfs) --------
// Loads a [tile_rows=128, tile_cols=32] f16 tile whose rows are strideK
// elements apart in memory into LDS at lds_off, inserting 4 dwords of padding
// after every 16 dwords (one 32-half row) -> LDS row stride = 40 halfs.
// rows_avail/cols_avail clamp reads (OOB returns zero per TDM spec).
__device__ __forceinline__ void tdm_load_tile(u32 lds_off, u64 gaddr,
                                              u32 rows_avail, u32 cols_avail,
                                              u32 strideK) {
  v4u g0;
  g0[0] = 1u;                                   // count=1, user descriptor
  g0[1] = lds_off;                              // LDS byte address
  g0[2] = (u32)gaddr;                           // global_addr[31:0]
  g0[3] = (u32)((gaddr >> 32) & 0x01ffffffull)  // global_addr[56:32]
        | (2u << 30);                           // type = 2 ("image")
  v8u g1;
  g1[0] = (1u << 16)    // data_size = 2 bytes
        | (1u << 20)    // pad_enable
        | (3u << 22)    // pad_interval: 16 dwords (2^(3+1))
        | (3u << 25);   // pad_amount: 4 dwords
  g1[1] = (cols_avail & 0xffffu) << 16;                         // tensor_dim0 lo
  g1[2] = (cols_avail >> 16) | ((rows_avail & 0xffffu) << 16);  // dim0 hi|dim1 lo
  g1[3] = (rows_avail >> 16) | (32u << 16);                     // dim1 hi|tile_dim0=32
  g1[4] = 128u;                                                 // tile_dim1=128, tile_dim2=0
  g1[5] = strideK;                                              // tensor_dim0_stride lo
  g1[6] = 0u;
  g1[7] = 0u;
  v4u gz;
  gz[0] = 0u; gz[1] = 0u; gz[2] = 0u; gz[3] = 0u;  // groups 2/3: dims unused
  asm volatile("tensor_load_to_lds %0, %1, %2, %3"
               :: "s"(g0), "s"(g1), "s"(gz), "s"(gz)
               : "memory");
}

// =============================== bookkeeping ===============================

__global__ void k_len(const int* __restrict__ bid, int* __restrict__ len, int T) {
  int t = blockIdx.x * 256 + threadIdx.x;
  if (t < T) atomicAdd(&len[bid[t]], 1);
}

__global__ __launch_bounds__(512) void k_scan(const int* __restrict__ len,
                                              int* __restrict__ offs) {
  __shared__ int s[512];
  int t = threadIdx.x;
  int v = len[t];
  s[t] = v; __syncthreads();
  for (int off = 1; off < 512; off <<= 1) {
    int u = (t >= off) ? s[t - off] : 0;
    __syncthreads();
    s[t] += u;
    __syncthreads();
  }
  offs[t] = s[t] - v;  // exclusive scan
}

__global__ void k_pos(const int* __restrict__ bid, const int* __restrict__ offs,
                      int* __restrict__ pos, int T) {
  int t = blockIdx.x * 256 + threadIdx.x;
  if (t < T) {
    int p = t - offs[bid[t]];
    pos[t] = p > (LL - 1) ? (LL - 1) : p;
  }
}

// =============================== converts ==================================

__global__ void k_f2h(const float* __restrict__ x, _Float16* __restrict__ y, int n) {
  int i = blockIdx.x * 256 + threadIdx.x;
  if (i < n) y[i] = (_Float16)x[i];
}

// transpose 128x128 f32 -> f16 (gcn_w is used as x @ W; our GEMM does x @ W^T)
__global__ __launch_bounds__(128) void k_gwt(const float* __restrict__ w,
                                             _Float16* __restrict__ wt) {
  int n = blockIdx.x, k = threadIdx.x;
  wt[n * DD + k] = (_Float16)w[k * DD + n];
}

__global__ void k_add(float* __restrict__ acc, const float* __restrict__ x, int n) {
  int i = blockIdx.x * 256 + threadIdx.x;
  if (i < n) acc[i] += x[i];
}

// =============================== gathers ===================================

__global__ __launch_bounds__(128)
void k_gather_seq(const float* __restrict__ poi, const int* __restrict__ tok,
                  const int* __restrict__ bid, const int* __restrict__ pos,
                  float* __restrict__ pad, _Float16* __restrict__ padh) {
  int t = blockIdx.x, d = threadIdx.x;
  int b = bid[t], p = pos[t];
  size_t dst = ((size_t)b * LL + p) * DD + d;
  float v = poi[(size_t)tok[t] * DD + d];
  pad[dst]  = v;
  padh[dst] = (_Float16)v;
}

__global__ __launch_bounds__(128)
void k_gather_loc(const _Float16* __restrict__ loch, const int* __restrict__ tok,
                  const int* __restrict__ bid, const int* __restrict__ pos,
                  _Float16* __restrict__ locpadh) {
  int t = blockIdx.x, d = threadIdx.x;
  int b = bid[t], p = pos[t];
  locpadh[((size_t)b * LL + p) * DD + d] = loch[(size_t)tok[t] * DD + d];
}

// =============================== layernorm =================================

__global__ __launch_bounds__(128)
void k_ln(const float* __restrict__ x, const float* __restrict__ g,
          const float* __restrict__ be, float* __restrict__ y,
          _Float16* __restrict__ yh, int ldh, int hoff) {
  int r = blockIdx.x, d = threadIdx.x;
  __shared__ float red[128];
  float v = x[(size_t)r * DD + d];
  red[d] = v; __syncthreads();
  for (int off = 64; off > 0; off >>= 1) {
    if (d < off) red[d] += red[d + off];
    __syncthreads();
  }
  float mean = red[0] * (1.0f / DD);
  __syncthreads();
  float c = v - mean;
  red[d] = c * c; __syncthreads();
  for (int off = 64; off > 0; off >>= 1) {
    if (d < off) red[d] += red[d + off];
    __syncthreads();
  }
  float var = red[0] * (1.0f / DD);
  float o = c * rsqrtf(var + 1e-8f) * g[d] + be[d];
  if (y)  y[(size_t)r * DD + d] = o;
  if (yh) yh[(size_t)r * ldh + hoff + d] = (_Float16)o;
}

// ====================== generic WMMA GEMM: X[M,K] @ W[N,K]^T ================
// block = 256 threads (8 waves). Block tile: 128(M) x 128(N), K tiled by 32.
// Wave w owns the 16-row strip across all 8 N-tiles (8 wmma per k-step).
// LDS tiles are filled by the Tensor Data Mover from wave 0 and double-
// buffered: the DMA for k-step t+1 overlaps the WMMAs of step t.
// ACT: 0 = none, 1 = leaky_relu(0.01).

template <int ACT>
__global__ __launch_bounds__(256)
void k_gemm(const _Float16* __restrict__ X, const _Float16* __restrict__ W,
            const float* __restrict__ bias, float* __restrict__ out,
            _Float16* __restrict__ outh, int M, int N, int K) {
  __shared__ _Float16 sX[2][128][40];
  __shared__ _Float16 sW[2][128][40];
  const int row0 = blockIdx.x * 128;
  const int col0 = blockIdx.y * 128;
  const int tid  = threadIdx.x;
  const int lane = tid & 31;
  const int wave = tid >> 5;
  const int nk   = K >> 5;
  const u32 TILE = 128u * 40u * 2u;  // bytes per LDS tile

  v8f acc[8];
#pragma unroll
  for (int i = 0; i < 8; i++)
#pragma unroll
    for (int j = 0; j < 8; j++) acc[i][j] = 0.0f;

  const u32 sx0 = (u32)(uintptr_t)&sX[0][0][0];
  const u32 sw0 = (u32)(uintptr_t)&sW[0][0][0];
  const u64 xg0 = (u64)(uintptr_t)X + (u64)row0 * (u64)K * 2ull;
  const u64 wg0 = (u64)(uintptr_t)W + (u64)col0 * (u64)K * 2ull;
  const u32 mrem = (u32)(M - row0);
  const u32 nrem = (u32)(N - col0);

  if (wave == 0) {  // prologue: DMA first k-tiles into buffer 0
    tdm_load_tile(sx0, xg0, mrem, (u32)K, (u32)K);
    tdm_load_tile(sw0, wg0, nrem, (u32)K, (u32)K);
    __builtin_amdgcn_s_wait_tensorcnt(0);
  }
  __syncthreads();

  for (int kt = 0; kt < nk; kt++) {
    const int cur = kt & 1;
    if (wave == 0 && (kt + 1) < nk) {  // prefetch next k-tile into other buffer
      const u32 nb = (u32)(cur ^ 1);
      const u32 k1 = (u32)(kt + 1) << 5;
      tdm_load_tile(sx0 + nb * TILE, xg0 + (u64)k1 * 2ull, mrem, (u32)K - k1, (u32)K);
      tdm_load_tile(sw0 + nb * TILE, wg0 + (u64)k1 * 2ull, nrem, (u32)K - k1, (u32)K);
    }
    const _Float16 (*cX)[40] = sX[cur];
    const _Float16 (*cW)[40] = sW[cur];

    // A fragment: 16x32 f16 per ISA layout (lanes 0-15: K {0..7,16..23};
    // lanes 16-31: K {8..15,24..31}); row m = lane&15 within wave's strip.
    v16h af;
    {
      int m = (lane & 15) + wave * 16;
      int koff = (lane >> 4) << 3;  // 0 or 8
#pragma unroll
      for (int i = 0; i < 16; i++) {
        int v = i >> 1, p = i & 1;
        int kk = (v < 4 ? 2 * v + p : 16 + 2 * (v - 4) + p) + koff;
        af[i] = cX[m][kk];
      }
    }
    // 8 N-tiles: B fragment 32x16 (lanes 0-15: K {0..7,16..23}; lanes 16-31: K {8..15,24..31})
#pragma unroll
    for (int nt = 0; nt < 8; nt++) {
      v16h bf;
      int n  = (lane & 15) + nt * 16;
      int kb = (lane >> 4) << 3;  // 0 or 8
#pragma unroll
      for (int i = 0; i < 16; i++) bf[i] = cW[n][kb + (i < 8 ? i : i + 8)];
      acc[nt] = __builtin_amdgcn_wmma_f32_16x16x32_f16(
          false, af, false, bf, (short)0, acc[nt], false, false);
    }
    if (wave == 0) __builtin_amdgcn_s_wait_tensorcnt(0);
    __syncthreads();
  }

  // store: C/D layout — VGPR r: lanes 0-15 M=r, lanes 16-31 M=8+r; N=lane&15
  int nl = lane & 15;
  int mb = (lane >> 4) << 3;
#pragma unroll
  for (int nt = 0; nt < 8; nt++) {
    int gn = col0 + nt * 16 + nl;
    if (gn < N) {
      float bv = bias ? bias[gn] : 0.0f;
#pragma unroll
      for (int r = 0; r < 8; r++) {
        int gm = row0 + wave * 16 + mb + r;
        if (gm < M) {
          float v = acc[nt][r] + bv;
          if (ACT == 1) v = v > 0.0f ? v : 0.01f * v;
          out[(size_t)gm * N + gn] = v;
          if (outh) outh[(size_t)gm * N + gn] = (_Float16)v;
        }
      }
    }
  }
}

// ======================= self attention (per b, per head) ===================
// blockDim=128: thread = query row. K/V head slices staged in LDS (f16),
// two-pass online softmax over valid keys (key_padding semantics).

__global__ __launch_bounds__(128)
void k_attn_self(const float* __restrict__ Qb,   // [B*L,128]
                 const float* __restrict__ KVb,  // [B*L,256] = K|V
                 const int* __restrict__ lens,
                 _Float16* __restrict__ attnh) { // [B*L,128]
  int b = blockIdx.x, h = blockIdx.y, q = threadIdx.x;
  __shared__ _Float16 sK[128][68];
  __shared__ _Float16 sV[128][68];
  size_t base = (size_t)b * LL;
  int len = lens[b];
  int kv = len < LL ? len : LL;

  for (int i = threadIdx.x; i < 128 * 64; i += 128) {
    int r = i >> 6, c = i & 63;
    sK[r][c] = (_Float16)KVb[(base + r) * 256 + h * 64 + c];
    sV[r][c] = (_Float16)KVb[(base + r) * 256 + 128 + h * 64 + c];
  }
  __syncthreads();

  _Float16 qr[64];
#pragma unroll
  for (int d = 0; d < 64; d++) qr[d] = (_Float16)Qb[(base + q) * 128 + h * 64 + d];

  // pass 1: running max + sum of exp
  float mmax = -1e30f, lsum = 0.0f;
  for (int k = 0; k < kv; k++) {
    float s = 0.0f;
#pragma unroll
    for (int d = 0; d < 64; d++) s += (float)qr[d] * (float)sK[k][d];
    s *= 0.125f;  // 1/sqrt(64)
    float mn = fmaxf(mmax, s);
    lsum = lsum * __expf(mmax - mn) + __expf(s - mn);
    mmax = mn;
  }
  float inv = 1.0f / lsum;

  // pass 2: weighted sum of V
  float o[64];
#pragma unroll
  for (int d = 0; d < 64; d++) o[d] = 0.0f;
  for (int k = 0; k < kv; k++) {
    float s = 0.0f;
#pragma unroll
    for (int d = 0; d < 64; d++) s += (float)qr[d] * (float)sK[k][d];
    float a = __expf(s * 0.125f - mmax) * inv;
#pragma unroll
    for (int d = 0; d < 64; d++) o[d] += a * (float)sV[k][d];
  }
#pragma unroll
  for (int d = 0; d < 64; d++) attnh[(base + q) * 128 + h * 64 + d] = (_Float16)o[d];
}

// ====================== residual + mask + ragged mean pool ==================

__global__ __launch_bounds__(128)
void k_pool(const float* __restrict__ proj, const float* __restrict__ qry,
            const int* __restrict__ lens, float* __restrict__ seqpre,
            _Float16* __restrict__ Acath) {
  int b = blockIdx.x, d = threadIdx.x;
  int len = lens[b];
  int n = len < LL ? len : LL;
  float s = 0.0f;
  for (int p = 0; p < n; p++) {
    size_t i = ((size_t)b * LL + p) * DD + d;
    s += proj[i] + qry[i];
  }
  float v = s / (float)len;
  seqpre[b * DD + d] = v;
  Acath[(size_t)b * 256 + d] = (_Float16)v;  // first half of concat(seq_pre, loc_pre)
}

// ====================== GCN edge scatter (weighted) ========================

__global__ __launch_bounds__(128)
void k_gcn_scatter(const float* __restrict__ x, const float* __restrict__ ew,
                   const int* __restrict__ src, const int* __restrict__ dst,
                   float* __restrict__ m) {
  int e = blockIdx.x, d = threadIdx.x;
  float w = ew[e];
  float v = w * x[(size_t)src[e] * DD + d];
  atomicAdd(&m[(size_t)dst[e] * DD + d], v);
}

// loc = acc/3; build f16 loc and the concatenated final weight [poi | loc]
__global__ __launch_bounds__(128)
void k_finalize_loc(const float* __restrict__ acc, const float* __restrict__ poi,
                    _Float16* __restrict__ loch, _Float16* __restrict__ Wcath) {
  int n = blockIdx.x, d = threadIdx.x;
  size_t i = (size_t)n * DD + d;
  float a = acc[i] * (1.0f / 3.0f);
  loch[i] = (_Float16)a;
  Wcath[(size_t)n * 256 + 128 + d] = (_Float16)a;
  Wcath[(size_t)n * 256 + d] = (_Float16)poi[i];
}

// ====================== cross attention (q_len = 1) ========================

__global__ __launch_bounds__(128)
void k_attn_cross(const float* __restrict__ Q2b,   // [B,128]
                  const float* __restrict__ KV2b,  // [B*L,256] = K|V
                  const int* __restrict__ lens,
                  _Float16* __restrict__ att2h) {  // [B,128]
  int b = blockIdx.x, tid = threadIdx.x;
  __shared__ float sQ[128], sA[128], red[128];
  sQ[tid] = Q2b[b * 128 + tid];
  int len = lens[b];
  int kv = len < LL ? len : LL;
  size_t base = (size_t)b * LL;
  __syncthreads();
  for (int h = 0; h < 2; h++) {
    float s = -1e30f;
    if (tid < kv) {
      s = 0.0f;
#pragma unroll
      for (int d = 0; d < 64; d++) s += sQ[h * 64 + d] * KV2b[(base + tid) * 256 + h * 64 + d];
      s *= 0.125f;
    }
    red[tid] = s; __syncthreads();
    for (int off = 64; off > 0; off >>= 1) {
      if (tid < off) red[tid] = fmaxf(red[tid], red[tid + off]);
      __syncthreads();
    }
    float mx = red[0]; __syncthreads();
    float e = (tid < kv) ? __expf(s - mx) : 0.0f;
    red[tid] = e; __syncthreads();
    for (int off = 64; off > 0; off >>= 1) {
      if (tid < off) red[tid] += red[tid + off];
      __syncthreads();
    }
    float inv = 1.0f / red[0]; __syncthreads();
    sA[tid] = e * inv; __syncthreads();
    if (tid < 64) {
      float o = 0.0f;
      for (int k = 0; k < kv; k++) o += sA[k] * KV2b[(base + k) * 256 + 128 + h * 64 + tid];
      att2h[b * 128 + h * 64 + tid] = (_Float16)o;
    }
    __syncthreads();
  }
}

// ================================ driver ===================================

static inline char* carve(char*& p, size_t bytes) {
  char* r = p;
  p += (bytes + 255) & ~(size_t)255;
  return r;
}

extern "C" void kernel_launch(void* const* d_in, const int* in_sizes, int n_in,
                              void* d_out, int out_size, void* d_ws, size_t ws_size,
                              hipStream_t stream) {
  (void)in_sizes; (void)n_in; (void)out_size; (void)ws_size;

  const float* poi       = (const float*)d_in[0];
  const float* gcn_w     = (const float*)d_in[1];
  const float* edge_w    = (const float*)d_in[2];
  const float* seq_ln_g  = (const float*)d_in[3];
  const float* seq_ln_b  = (const float*)d_in[4];
  const float* seq_in_w  = (const float*)d_in[5];
  const float* seq_in_b  = (const float*)d_in[6];
  const float* seq_out_w = (const float*)d_in[7];
  const float* seq_out_b = (const float*)d_in[8];
  const float* geo_ln_g  = (const float*)d_in[9];
  const float* geo_ln_b  = (const float*)d_in[10];
  const float* geo_in_w  = (const float*)d_in[11];
  const float* geo_in_b  = (const float*)d_in[12];
  const float* geo_out_w = (const float*)d_in[13];
  const float* geo_out_b = (const float*)d_in[14];
  const float* go_ln_g   = (const float*)d_in[15];
  const float* go_ln_b   = (const float*)d_in[16];
  const int*   e_src     = (const int*)d_in[17];
  const int*   e_dst     = (const int*)d_in[18];
  const int*   tok       = (const int*)d_in[19];
  const int*   bid       = (const int*)d_in[20];
  float* out = (float*)d_out;

  const size_t TOK = (size_t)BB * LL;  // 65536 padded token rows

  char* p = (char*)d_ws;
  int*      lens    = (int*)carve(p, BB * 4);
  int*      offs    = (int*)carve(p, BB * 4);
  int*      pos     = (int*)carve(p, TT * 4);
  float*    pad     = (float*)carve(p, TOK * DD * 4);
  _Float16* padh    = (_Float16*)carve(p, TOK * DD * 2);
  float*    qry     = (float*)carve(p, TOK * DD * 4);
  _Float16* qryh    = (_Float16*)carve(p, TOK * DD * 2);
  float*    Qbuf    = (float*)carve(p, TOK * DD * 4);     // reused as `proj`
  float*    KVbuf   = (float*)carve(p, TOK * 256 * 4);    // reused as KV2
  _Float16* attnh   = (_Float16*)carve(p, TOK * DD * 2);
  float*    seqpre  = (float*)carve(p, BB * DD * 4);
  _Float16* q2h     = (_Float16*)carve(p, BB * DD * 2);
  float*    Q2buf   = (float*)carve(p, BB * DD * 4);
  _Float16* att2h   = (_Float16*)carve(p, BB * DD * 2);
  float*    att2p   = (float*)carve(p, BB * DD * 4);
  float*    mbuf    = (float*)carve(p, (size_t)NPOI * DD * 4);
  _Float16* mh      = (_Float16*)carve(p, (size_t)NPOI * DD * 2);
  float*    xbuf    = (float*)carve(p, (size_t)NPOI * DD * 4);
  float*    accb    = (float*)carve(p, (size_t)NPOI * DD * 4);
  _Float16* loch    = (_Float16*)carve(p, (size_t)NPOI * DD * 2);
  _Float16* locpadh = (_Float16*)carve(p, TOK * DD * 2);
  _Float16* siwh    = (_Float16*)carve(p, 384 * DD * 2);
  _Float16* sowh    = (_Float16*)carve(p, DD * DD * 2);
  _Float16* giwh    = (_Float16*)carve(p, 384 * DD * 2);
  _Float16* gowh    = (_Float16*)carve(p, DD * DD * 2);
  _Float16* gwt0    = (_Float16*)carve(p, DD * DD * 2);
  _Float16* gwt1    = (_Float16*)carve(p, DD * DD * 2);
  _Float16* Acath   = (_Float16*)carve(p, BB * 256 * 2);
  _Float16* Wcath   = (_Float16*)carve(p, (size_t)NPOI * 256 * 2);

  // ---- bookkeeping ----
  hipMemsetAsync(lens, 0, BB * 4, stream);
  k_len<<<(TT + 255) / 256, 256, 0, stream>>>(bid, lens, TT);
  k_scan<<<1, 512, 0, stream>>>(lens, offs);
  k_pos<<<(TT + 255) / 256, 256, 0, stream>>>(bid, offs, pos, TT);

  // ---- weight conversions ----
  k_f2h<<<(384 * DD + 255) / 256, 256, 0, stream>>>(seq_in_w, siwh, 384 * DD);
  k_f2h<<<(DD * DD + 255) / 256, 256, 0, stream>>>(seq_out_w, sowh, DD * DD);
  k_f2h<<<(384 * DD + 255) / 256, 256, 0, stream>>>(geo_in_w, giwh, 384 * DD);
  k_f2h<<<(DD * DD + 255) / 256, 256, 0, stream>>>(geo_out_w, gowh, DD * DD);
  k_gwt<<<DD, DD, 0, stream>>>(gcn_w, gwt0);
  k_gwt<<<DD, DD, 0, stream>>>(gcn_w + DD * DD, gwt1);

  // ---- sequence_Pro ----
  hipMemsetAsync(pad, 0, TOK * DD * 4, stream);
  hipMemsetAsync(padh, 0, TOK * DD * 2, stream);
  k_gather_seq<<<TT, DD, 0, stream>>>(poi, tok, bid, pos, pad, padh);
  k_ln<<<(int)TOK, DD, 0, stream>>>(pad, seq_ln_g, seq_ln_b, qry, qryh, DD, 0);

  dim3 gQ((TOK + 127) / 128, 1);
  k_gemm<0><<<gQ, 256, 0, stream>>>(qryh, siwh, seq_in_b, Qbuf, nullptr,
                                    (int)TOK, DD, DD);
  dim3 gKV((TOK + 127) / 128, 2);
  k_gemm<0><<<gKV, 256, 0, stream>>>(padh, siwh + 128 * DD, seq_in_b + 128, KVbuf,
                                     nullptr, (int)TOK, 256, DD);
  k_attn_self<<<dim3(BB, 2), 128, 0, stream>>>(Qbuf, KVbuf, lens, attnh);
  float* proj = Qbuf;  // Q dead after attention
  k_gemm<0><<<gQ, 256, 0, stream>>>(attnh, sowh, seq_out_b, proj, nullptr,
                                    (int)TOK, DD, DD);
  k_pool<<<BB, DD, 0, stream>>>(proj, qry, lens, seqpre, Acath);

  // ---- GeoEncoder (2-layer weighted GCN) ----
  hipMemcpyAsync(accb, poi, (size_t)NPOI * DD * 4, hipMemcpyDeviceToDevice, stream);
  dim3 gGC((NPOI + 127) / 128, 1);
  // layer 0
  hipMemsetAsync(mbuf, 0, (size_t)NPOI * DD * 4, stream);
  k_gcn_scatter<<<EE, DD, 0, stream>>>(poi, edge_w, e_src, e_dst, mbuf);
  k_f2h<<<((int)((size_t)NPOI * DD) + 255) / 256, 256, 0, stream>>>(mbuf, mh, NPOI * DD);
  k_gemm<1><<<gGC, 256, 0, stream>>>(mh, gwt0, nullptr, xbuf, nullptr, NPOI, DD, DD);
  k_add<<<((int)((size_t)NPOI * DD) + 255) / 256, 256, 0, stream>>>(accb, xbuf, NPOI * DD);
  // layer 1
  hipMemsetAsync(mbuf, 0, (size_t)NPOI * DD * 4, stream);
  k_gcn_scatter<<<EE, DD, 0, stream>>>(xbuf, edge_w, e_src, e_dst, mbuf);
  k_f2h<<<((int)((size_t)NPOI * DD) + 255) / 256, 256, 0, stream>>>(mbuf, mh, NPOI * DD);
  k_gemm<1><<<gGC, 256, 0, stream>>>(mh, gwt1, nullptr, xbuf, nullptr, NPOI, DD, DD);
  k_add<<<((int)((size_t)NPOI * DD) + 255) / 256, 256, 0, stream>>>(accb, xbuf, NPOI * DD);
  k_finalize_loc<<<NPOI, DD, 0, stream>>>(accb, poi, loch, Wcath);

  // ---- location_Pro ----
  hipMemsetAsync(locpadh, 0, TOK * DD * 2, stream);
  k_gather_loc<<<TT, DD, 0, stream>>>(loch, tok, bid, pos, locpadh);
  float* KV2 = KVbuf;  // seq KV dead
  k_gemm<0><<<gKV, 256, 0, stream>>>(locpadh, giwh + 128 * DD, geo_in_b + 128, KV2,
                                     nullptr, (int)TOK, 256, DD);
  k_ln<<<BB, DD, 0, stream>>>(seqpre, geo_ln_g, geo_ln_b, nullptr, q2h, DD, 0);
  dim3 gS((BB + 127) / 128, 1);
  k_gemm<0><<<gS, 256, 0, stream>>>(q2h, giwh, geo_in_b, Q2buf, nullptr, BB, DD, DD);
  k_attn_cross<<<BB, 128, 0, stream>>>(Q2buf, KV2, lens, att2h);
  k_gemm<0><<<gS, 256, 0, stream>>>(att2h, gowh, geo_out_b, att2p, nullptr, BB, DD, DD);
  k_ln<<<BB, DD, 0, stream>>>(att2p, go_ln_g, go_ln_b, nullptr, Acath, 256, 128);

  // ---- final lookup GEMM: [512,256] x [100000,256]^T -> [512,100000] ----
  dim3 gF((BB + 127) / 128, (NPOI + 127) / 128);
  k_gemm<0><<<gF, 256, 0, stream>>>(Acath, Wcath, nullptr, out, nullptr,
                                    BB, NPOI, 256);
}
